// KANLayer_20615843021187
// MI455X (gfx1250) — matrix operand order repaired and checked by reading, not compiled, not run on hardware
//
#include <hip/hip_runtime.h>

// ---------------------------------------------------------------------------
// KAN layer for MI455X (gfx1250, wave32, WMMA).
//
//   out[b,o] = r[b] + sum_k DM[b,k] * W[k,o]
//     DM[b, i*8+c] = JacobiP_c(tanh(x[b,i]))            (A: 4096 x 2048, f16)
//     W [i*8+c, o] = spline_scale[i,o]*coefs[i,o,c]/256 (B: 2048 x 256, f16,
//                    transposed [o][k] in workspace)
//     r[b]         = (1/256) sum_i resid_scale[i]*tanh(x[b,i])
//
// GEMM = 4.3 GFLOP vs ~10 MB HBM traffic -> compute-limited; f16 operands
// (v_wmma_f32_16x16x32_f16, f32 accumulation) give 8x fewer matrix ops than
// the f32 16x16x4 path; inputs are bounded (tanh, Jacobi O(1), coefs ~0.03).
//
// Final structure (round 4): builtin WMMA (NOT inline asm) so the compiler
// inserts the mandatory WMMA->VALU hazard v_nops around its own accumulator
// moves -- the asm-tied variant pipelined slightly deeper but left
// compiler-generated v_mov_b64 reads of WMMA D inside the 4-coexec hazard
// window with no protection. The residual per-iteration accumulator moves
// issue on the plain VALU pipe and co-execute under the XDL WMMAs (ISA 8.4),
// so they are an acceptable price for hazard-correct code.
//   * wave tile 32Mx32N (2x2 WMMA tiles): 4 WMMAs per 2 A-frags + 2 B-frags;
//     the 1 MB W matrix is read exactly once per block (L2-resident).
//   * 2-deep register double-buffering: step k+1's LDS/global loads are in
//     flight behind step k's 4 WMMAs (disasm: partial s_wait_loadcnt/dscnt).
// ---------------------------------------------------------------------------

typedef _Float16 v16h __attribute__((ext_vector_type(16)));
typedef _Float16 v8h  __attribute__((ext_vector_type(8)));
typedef float    v8f  __attribute__((ext_vector_type(8)));

#define BATCH 4096
#define INDIM 256
#define OUTDIM 256
#define NC    8
#define KDIM  (INDIM * NC)   // 2048
#define BM    32             // batch rows per block
#define ROWH  (KDIM + 8)     // padded LDS row stride in halves (+16B -> 4-bank step)

union F16x16 { v16h v; v8h h[2]; };

// ---- Stage 1: fold spline_scale & 1/IN into f16 B-matrix, transposed [o][k]
__global__ __launch_bounds__(256) void kan_prep_w(
    const float* __restrict__ coefs,         // [IN][OUT][NC]
    const float* __restrict__ spline_scale,  // [IN][OUT]
    _Float16* __restrict__ wt)               // [OUT][KDIM]
{
  const int i = blockIdx.x;    // input index
  const int o = threadIdx.x;   // output index
  const float* cp = coefs + ((size_t)i * OUTDIM + o) * NC;   // 32B contiguous/lane
  const float ss = spline_scale[i * OUTDIM + o] * (1.0f / (float)INDIM);
  _Float16* dst = wt + (size_t)o * KDIM + i * NC;            // 16B aligned
  v8h hv;
#pragma unroll
  for (int c = 0; c < NC; ++c) hv[c] = (_Float16)(cp[c] * ss);
  *(v8h*)dst = hv;
}

// ---- Stage 2: DM build (LDS) + WMMA GEMM + residual rank-1 add
__global__ __launch_bounds__(256) void kan_wmma(
    const float* __restrict__ x,             // [B][IN]
    const float* __restrict__ alpha_arctanh, // scalar
    const float* __restrict__ resid_scale,   // [IN]
    const _Float16* __restrict__ wt,         // [OUT][KDIM]
    float* __restrict__ out)                 // [B][OUT]
{
  extern __shared__ __align__(16) char smem[];
  _Float16* dm   = (_Float16*)smem;                    // [BM][ROWH]
  float*    rrow = (float*)(smem + (size_t)BM * ROWH * sizeof(_Float16)); // [BM]

  const int tid  = threadIdx.x;
  const int lane = tid & 31;
  const int w    = tid >> 5;           // wave id 0..7
  const int row0 = blockIdx.x * BM;

  const float alpha = tanhf(alpha_arctanh[0]);
  // Jacobi three-term recurrence constants (alpha == beta -> a^2-b^2 term = 0)
  float bco[NC], cco[NC], ain[NC];
#pragma unroll
  for (int n = 2; n < NC; ++n) {
    const float c = 2.0f * n + 2.0f * alpha;
    bco[n] = (c - 1.0f) * c * (c - 2.0f);
    cco[n] = 2.0f * (n + alpha - 1.0f) * (n + alpha - 1.0f) * c;
    ain[n] = 1.0f / (2.0f * n * (n + 2.0f * alpha) * (c - 2.0f));
  }

  // ---------- Phase 1: design matrix tile -> LDS (f16) + residual sums ------
#pragma unroll
  for (int rr = 0; rr < BM / 8; ++rr) {
    const int r = w * (BM / 8) + rr;                 // block-local batch row
    const float* xr = x + (size_t)(row0 + r) * INDIM;
    float s = 0.0f;
#pragma unroll
    for (int j = 0; j < INDIM / 32; ++j) {
      const int i = lane + 32 * j;                   // coalesced x load
      const float t = tanhf(xr[i]);
      s += resid_scale[i] * t;
      float p0 = 1.0f;
      float p1 = (alpha + 1.0f) + (alpha + alpha + 2.0f) * (t - 1.0f) * 0.5f;
      v8h hv;
      hv[0] = (_Float16)p0;
      hv[1] = (_Float16)p1;
#pragma unroll
      for (int n = 2; n < NC; ++n) {
        const float pn = (bco[n] * t * p1 - cco[n] * p0) * ain[n];
        hv[n] = (_Float16)pn;
        p0 = p1; p1 = pn;
      }
      *(v8h*)(dm + (size_t)r * ROWH + i * NC) = hv;  // 16B LDS store, no conflict
    }
    // wave32 reduction of residual partial
#pragma unroll
    for (int off = 16; off > 0; off >>= 1) s += __shfl_xor(s, off, 32);
    if (lane == 0) rrow[r] = s * (1.0f / (float)INDIM);
  }
  __syncthreads();

  // ---------- Phase 2: WMMA GEMM, wave -> 32(M) x 32(N) tile ----------------
  // 8 waves each own a 32-column slice of the 32x256 block tile.
  const int kbA = (lane >> 4) * 8;      // A-frag K sub-offset (ISA 16-bit A layout)
  const int kbB = (lane >> 4) * 8;      // B-frag K sub-offset (ISA 16-bit B layout)
  const _Float16* aptr[2] = {
      dm + (size_t)(lane & 15) * ROWH,
      dm + (size_t)(16 + (lane & 15)) * ROWH };
  const _Float16* bptr[2] = {
      wt + (size_t)(w * 32 +      (lane & 15)) * KDIM + kbB,
      wt + (size_t)(w * 32 + 16 + (lane & 15)) * KDIM + kbB };

  v8f acc[2][2] = {};
  F16x16 a[2][2], b[2][2];   // [buffer][tile] -- only literal buffer indices!

  auto load_frags = [&](int buf, int ka) {
#pragma unroll
    for (int m = 0; m < 2; ++m) {
      a[buf][m].h[0] = *(const v8h*)(aptr[m] + ka + kbA);        // K=ka+kb..+7
      a[buf][m].h[1] = *(const v8h*)(aptr[m] + ka + 16 + kbA);   // K=ka+16+kb..+7
    }
#pragma unroll
    for (int nf = 0; nf < 2; ++nf) {
      b[buf][nf].h[0] = *(const v8h*)(bptr[nf] + ka);            // K=ka+kbB..+7
      b[buf][nf].h[1] = *(const v8h*)(bptr[nf] + ka + 16);       // K=ka+16+kbB..+7
    }
  };
  auto do_wmma = [&](int buf) {
#pragma unroll
    for (int m = 0; m < 2; ++m)
#pragma unroll
      for (int nf = 0; nf < 2; ++nf)
        acc[m][nf] = __builtin_amdgcn_wmma_f32_16x16x32_f16(
            false, a[buf][m].v, false, b[buf][nf].v, (short)0,
            acc[m][nf], false, false);
  };

  // Software pipeline: loads for step k+1 in flight behind step k's 4 WMMAs.
  load_frags(0, 0);
  for (int ks = 0; ks < 62; ks += 2) {
    load_frags(1, (ks + 1) * 32);
    do_wmma(0);
    load_frags(0, (ks + 2) * 32);
    do_wmma(1);
  }
  load_frags(1, 63 * 32);
  do_wmma(0);
  do_wmma(1);

  // ---------- Writeback: C/D layout VGPR j -> M=j+8*(lane>>4), N=lane&15 ----
  const int rb = (lane >> 4) * 8;
  const int oc = w * 32 + (lane & 15);
#pragma unroll
  for (int m = 0; m < 2; ++m)
#pragma unroll
    for (int nf = 0; nf < 2; ++nf)
#pragma unroll
      for (int j = 0; j < 8; ++j) {
        const int rl = m * 16 + rb + j;             // block-local batch row
        out[(size_t)(row0 + rl) * OUTDIM + oc + nf * 16] =
            acc[m][nf][j] + rrow[rl];
      }
}

extern "C" void kernel_launch(void* const* d_in, const int* in_sizes, int n_in,
                              void* d_out, int out_size, void* d_ws, size_t ws_size,
                              hipStream_t stream) {
  const float* x    = (const float*)d_in[0];  // [4096,256]
  const float* cf   = (const float*)d_in[1];  // [256,256,8]
  const float* aat  = (const float*)d_in[2];  // scalar
  const float* rs   = (const float*)d_in[3];  // [256,1]
  const float* ssc  = (const float*)d_in[4];  // [256,256]
  float* out = (float*)d_out;                 // [4096,256] f32
  _Float16* wt = (_Float16*)d_ws;             // 256*2048 halves = 1 MiB scratch

  // Stage 1: build folded f16 B-matrix (transposed [o][k]).
  kan_prep_w<<<INDIM, OUTDIM, 0, stream>>>(cf, ssc, wt);

  // Stage 2: 128 blocks x 8 waves; ~128.6 KB dynamic LDS for the DM tile.
  const size_t smem = (size_t)BM * ROWH * sizeof(_Float16) + BM * sizeof(float);
  kan_wmma<<<BATCH / BM, 256, smem, stream>>>(x, aat, rs, wt, out);
}
